// MambaLM_20057497272773
// MI455X (gfx1250) — matrix operand order repaired and checked by reading, not compiled, not run on hardware
//
#include <hip/hip_runtime.h>
#include <hip/hip_bf16.h>

// ---------------- model constants ----------------
#define Bn   16
#define Ln   512
#define Dn   512
#define Vn   10000
#define NLn  4
#define EDn  1024
#define Nst  16
#define DTRn 32
#define Mrows (Bn*Ln)            // 8192

typedef __attribute__((ext_vector_type(16))) __bf16 bf16x16;
typedef __attribute__((ext_vector_type(8)))  __bf16 bf16x8;
typedef __attribute__((ext_vector_type(8)))  float  f32x8;

// pointee/addrspace types for the async global->LDS DMA builtin:
// param0: int4 in AS1 (global/device), param1: int4 in AS3 (shared/LDS)
typedef int v4i_t __attribute__((vector_size(16)));
typedef __attribute__((address_space(1))) v4i_t glb_v4i;
typedef __attribute__((address_space(3))) v4i_t lds_v4i;

union FragU { bf16x16 v; bf16x8 h[2]; };

enum { EPI_NONE = 0, EPI_SIGB = 1, EPI_SOFTPLUSB = 2, EPI_RESID = 3 };

// Async global->LDS DMA path (gfx1250, ASYNCcnt-tracked) when the toolchain
// exposes it; otherwise fall back to load->VGPR->ds_store staging.
#if defined(__gfx1250__) && __has_builtin(__builtin_amdgcn_global_load_async_to_lds_b128)
#define ASYNC_LDS 1
#else
#define ASYNC_LDS 0
#endif

#if ASYNC_LDS
// clang rejects direct pointer casts across address spaces; round-trip through
// integers (CK-style). Low 32 bits of a flat shared pointer == LDS offset.
__device__ __forceinline__ glb_v4i* to_glb_v4i(const void* p) {
  return (glb_v4i*)(unsigned long long)p;
}
__device__ __forceinline__ lds_v4i* to_lds_v4i(const void* p) {
  return (lds_v4i*)(unsigned int)(unsigned long long)p;
}
#endif

__device__ __forceinline__ void wait_async_lds() {
#if ASYNC_LDS
#if __has_builtin(__builtin_amdgcn_s_wait_asynccnt)
  __builtin_amdgcn_s_wait_asynccnt(0);
#else
  asm volatile("s_wait_asynccnt 0x0" ::: "memory");
#endif
#endif
}

// stage one 128x32 A tile + 128x32 B tile (bf16) into LDS (stride 48)
__device__ __forceinline__ void stage_tile(const __bf16* __restrict__ A,
                                           const __bf16* __restrict__ Bw,
                                           __bf16* sA, __bf16* sB,
                                           int tid, int m0, int n0,
                                           int N, int K, int k0) {
  for (int i = 0; i < 2; ++i) {
    int j   = tid + i * 256;       // 0..511
    int row = j >> 2;              // 0..127
    int col = (j & 3) << 3;        // 0,8,16,24 (bf16 units)
    const __bf16* ga = A + (size_t)(m0 + row) * K + (k0 + col);
    __bf16*       la = &sA[row * 48 + col];
    int nr = n0 + row;
    const __bf16* gb = Bw + (size_t)nr * K + (k0 + col);
    __bf16*       lb = &sB[row * 48 + col];
#if ASYNC_LDS
    __builtin_amdgcn_global_load_async_to_lds_b128(to_glb_v4i(ga), to_lds_v4i(la), 0, 0);
    if (nr < N) {
      __builtin_amdgcn_global_load_async_to_lds_b128(to_glb_v4i(gb), to_lds_v4i(lb), 0, 0);
    } else {
      bf16x8 z;
      for (int q = 0; q < 8; ++q) z[q] = (__bf16)0.0f;
      *(bf16x8*)lb = z;
    }
#else
    *(bf16x8*)la = *(const bf16x8*)ga;
    bf16x8 bv;
    if (nr < N) {
      bv = *(const bf16x8*)gb;
    } else {
      for (int q = 0; q < 8; ++q) bv[q] = (__bf16)0.0f;
    }
    *(bf16x8*)lb = bv;
    if (k0 + 32 < K) {             // prefetch next K tile -> global_prefetch_b8
      __builtin_prefetch(ga + 32, 0, 1);
      if (nr < N) __builtin_prefetch(gb + 32, 0, 1);
    }
#endif
  }
}

// =======================================================================
// Tiled WMMA GEMM:  C[M,N] = epi( A[M,K] @ B[N,K]^T )
// A,B row-major bf16; f32 accumulation via v_wmma_f32_16x16x32_bf16.
// Block tile 128x128, K-step 32, 8 waves (wave32), wave tile 32x64.
// Double-buffered LDS; async DMA staging overlapped with WMMA compute.
// =======================================================================
template<typename TOUT, int EPI>
__global__ __launch_bounds__(256)
void gemm_wmma_bf16(const __bf16* __restrict__ A, const __bf16* __restrict__ Bw,
                    TOUT* __restrict__ C, const float* __restrict__ bias,
                    const float* __restrict__ resid, int M, int N, int K) {
  __shared__ alignas(16) __bf16 sA[2][128 * 48];   // 48 = 32 + pad, keeps 16B align
  __shared__ alignas(16) __bf16 sB[2][128 * 48];

  const int tid  = threadIdx.x;
  const int lane = tid & 31;
  const int wv   = tid >> 5;
  const int wm   = wv >> 1;          // 0..3  (M direction)
  const int wn   = wv & 1;           // 0..1  (N direction)
  const int m0   = blockIdx.y * 128;
  const int n0   = blockIdx.x * 128;
  const int mrow = lane & 15;
  const int kh   = lane >> 4;        // lane half selects K sub-range

  f32x8 acc[2][4];
  for (int mi = 0; mi < 2; ++mi)
    for (int ni = 0; ni < 4; ++ni)
      for (int q = 0; q < 8; ++q) acc[mi][ni][q] = 0.0f;

  stage_tile(A, Bw, sA[0], sB[0], tid, m0, n0, N, K, 0);
  wait_async_lds();
  __syncthreads();

  int buf = 0;
  for (int k0 = 0; k0 < K; k0 += 32) {
    // kick off next tile's DMA while we compute on the current one
    if (k0 + 32 < K)
      stage_tile(A, Bw, sA[buf ^ 1], sB[buf ^ 1], tid, m0, n0, N, K, k0 + 32);

    // ---- load WMMA fragments from LDS (ISA 16-bit A/B lane layouts) ----
    bf16x16 af[2], bfr[4];
    for (int mi = 0; mi < 2; ++mi) {
      int r = wm * 32 + mi * 16 + mrow;
      FragU u;
      u.h[0] = *(const bf16x8*)&sA[buf][r * 48 + kh * 8];        // K = kh*8 .. +7
      u.h[1] = *(const bf16x8*)&sA[buf][r * 48 + 16 + kh * 8];   // K = 16+kh*8 .. +7
      af[mi] = u.v;
    }
    for (int ni = 0; ni < 4; ++ni) {
      int r = wn * 64 + ni * 16 + mrow;
      FragU u;
      u.h[0] = *(const bf16x8*)&sB[buf][r * 48 + kh * 8];        // K = kh*8 .. +7
      u.h[1] = *(const bf16x8*)&sB[buf][r * 48 + 16 + kh * 8];   // K = 16+kh*8 .. +7
      bfr[ni] = u.v;
    }

    for (int mi = 0; mi < 2; ++mi)
      for (int ni = 0; ni < 4; ++ni)
        acc[mi][ni] = __builtin_amdgcn_wmma_f32_16x16x32_bf16(
            false, af[mi], false, bfr[ni], (short)0, acc[mi][ni], false, false);

    wait_async_lds();   // this wave's next-tile DMA done before everyone flips
    __syncthreads();
    buf ^= 1;
  }

  // ---- epilogue + store (C/D layout: VGPR r -> M = r + 8*lanehalf) ----
  for (int mi = 0; mi < 2; ++mi)
    for (int ni = 0; ni < 4; ++ni) {
      int col = n0 + wn * 64 + ni * 16 + mrow;
      if (col >= N) continue;
      float bcol = (EPI == EPI_SIGB || EPI == EPI_SOFTPLUSB) ? bias[col] : 0.0f;
      for (int r = 0; r < 8; ++r) {
        int row = m0 + wm * 32 + mi * 16 + r + (kh << 3);
        float v = acc[mi][ni][r];
        if (EPI == EPI_SIGB) {
          v = 1.0f / (1.0f + __expf(-(v + bcol)));
        } else if (EPI == EPI_SOFTPLUSB) {
          float t = v + bcol;
          v = (t > 20.0f) ? t : log1pf(__expf(t));
        } else if (EPI == EPI_RESID) {
          v = v + resid[(size_t)row * N + col];
        }
        C[(size_t)row * N + col] = (TOUT)v;
      }
    }
}

// =======================================================================
// small helper kernels
// =======================================================================
__global__ void cvt_bf16_kernel(const float* __restrict__ s, __bf16* __restrict__ d, size_t n) {
  size_t i = (size_t)blockIdx.x * blockDim.x + threadIdx.x;
  if (i < n) d[i] = (__bf16)s[i];
}

__global__ void copy_f32_kernel(const float* __restrict__ s, float* __restrict__ d, size_t n) {
  size_t i = (size_t)blockIdx.x * blockDim.x + threadIdx.x;
  if (i < n) d[i] = s[i];
}

__global__ void neg_exp_kernel(const float* __restrict__ s, float* __restrict__ d, size_t n) {
  size_t i = (size_t)blockIdx.x * blockDim.x + threadIdx.x;
  if (i < n) d[i] = -__expf(s[i]);
}

__global__ void slice_cvt_kernel(const float* __restrict__ s, __bf16* __restrict__ d,
                                 int rows, int scols, int dcols) {
  size_t i = (size_t)blockIdx.x * blockDim.x + threadIdx.x;
  if (i >= (size_t)rows * dcols) return;
  int r = (int)(i / dcols), c = (int)(i % dcols);
  d[i] = (__bf16)s[(size_t)r * scols + c];
}

__global__ void enc_mean_kernel(const float* __restrict__ E, float* __restrict__ enc) {
  int i = blockIdx.x * blockDim.x + threadIdx.x;     // B*D
  if (i >= Bn * Dn) return;
  int b = i / Dn, d = i % Dn;
  const float* p = E + (size_t)b * 3 * 49 * Dn + d;  // EncoderOut[b,0,:,d]
  float s = 0.0f;
  for (int j = 0; j < 49; ++j) s += p[(size_t)j * Dn];
  enc[i] = s * (1.0f / 49.0f);
}

__global__ void build_xe_kernel(const int* __restrict__ tok, const float* __restrict__ emb,
                                const float* __restrict__ enc, __bf16* __restrict__ xe) {
  size_t i = (size_t)blockIdx.x * blockDim.x + threadIdx.x;   // M*D
  if (i >= (size_t)Mrows * Dn) return;
  size_t m = i / Dn; int d = (int)(i % Dn);
  int b = (int)(m / Ln);
  int t = tok[m];
  xe[m * (2 * Dn) + d]      = (__bf16)emb[(size_t)t * Dn + d];
  xe[m * (2 * Dn) + Dn + d] = (__bf16)enc[(size_t)b * Dn + d];
}

__global__ void gates_combine_kernel(const float* __restrict__ G, const float* __restrict__ enc,
                                     float* __restrict__ h) {
  size_t i = (size_t)blockIdx.x * blockDim.x + threadIdx.x;   // M*D
  if (i >= (size_t)Mrows * Dn) return;
  size_t m = i / Dn; int d = (int)(i % Dn);
  int b = (int)(m / Ln);
  float e = enc[(size_t)b * Dn + d];
  float g = G[m * 1536 + d] + G[m * 1536 + 512 + d] + G[m * 1536 + 1024 + d];
  h[i] = e * g * 0.57735026919f;   // 1/sqrt(3)
}

__global__ __launch_bounds__(256)
void rmsnorm_kernel(const float* __restrict__ x, const float* __restrict__ w,
                    __bf16* __restrict__ out) {
  __shared__ float red[256];
  size_t row = blockIdx.x;
  const float* xr = x + row * Dn;
  int tid = threadIdx.x;
  float v0 = xr[tid], v1 = xr[tid + 256];
  red[tid] = v0 * v0 + v1 * v1;
  __syncthreads();
  for (int s = 128; s > 0; s >>= 1) {
    if (tid < s) red[tid] += red[tid + s];
    __syncthreads();
  }
  float rs = rsqrtf(red[0] * (1.0f / Dn) + 1e-5f);
  out[row * Dn + tid]       = (__bf16)(v0 * rs * w[tid]);
  out[row * Dn + tid + 256] = (__bf16)(v1 * rs * w[tid + 256]);
}

// depthwise causal conv (K=4) + bias + SiLU; emits f32 (for scan) and bf16 (for GEMM)
__global__ void conv_silu_kernel(const float* __restrict__ xz, const float* __restrict__ w,
                                 const float* __restrict__ bias,
                                 float* __restrict__ xic, __bf16* __restrict__ xib) {
  size_t i = (size_t)blockIdx.x * blockDim.x + threadIdx.x;   // M*ED
  if (i >= (size_t)Mrows * EDn) return;
  int e = (int)(i % EDn);
  size_t m = i / EDn;
  int l = (int)(m % Ln);
  const float* wr = w + (size_t)e * 4;
  float s = bias[e];
  for (int k = 0; k < 4; ++k) {
    int ll = l - 3 + k;
    if (ll >= 0) s += wr[k] * xz[(m - 3 + k) * (2 * EDn) + e];
  }
  float v = s / (1.0f + __expf(-s));   // silu
  xic[i] = v;
  xib[i] = (__bf16)v;
}

// selective scan: one block per batch, one thread per channel, state in regs.
__global__ __launch_bounds__(1024)
void scan_kernel(const float* __restrict__ dbc, const float* __restrict__ delta,
                 const float* __restrict__ xi, const float* __restrict__ xz,
                 const float* __restrict__ Aneg, const float* __restrict__ Dp,
                 __bf16* __restrict__ ygate) {
  __shared__ float bc[2][32];        // [Bm(16) | Cm(16)], double buffered
  int b = blockIdx.x;
  int e = threadIdx.x;
  float h[Nst], a[Nst];
  for (int n = 0; n < Nst; ++n) { h[n] = 0.0f; a[n] = Aneg[(size_t)e * Nst + n]; }
  float dp = Dp[e];
  for (int t = 0; t < Ln; ++t) {
    size_t m = (size_t)b * Ln + t;
    int buf = t & 1;
    if (e < 32) bc[buf][e] = dbc[m * 64 + 32 + e];
    __syncthreads();
    float d  = delta[m * EDn + e];
    float x  = xi[m * EDn + e];
    float dx = d * x;
    float y  = 0.0f;
    for (int n = 0; n < Nst; ++n) {
      h[n] = __expf(d * a[n]) * h[n] + dx * bc[buf][n];
      y += h[n] * bc[buf][16 + n];
    }
    float z = xz[m * (2 * EDn) + EDn + e];
    float g = (y + dp * x) * (z / (1.0f + __expf(-z)));
    ygate[m * EDn + e] = (__bf16)g;
  }
}

__global__ __launch_bounds__(256)
void logsoftmax_kernel(float* __restrict__ x, int n) {
  __shared__ float red[256];
  float* xr = x + (size_t)blockIdx.x * n;
  int tid = threadIdx.x;
  float mx = -3.4e38f;
  for (int j = tid; j < n; j += 256) mx = fmaxf(mx, xr[j]);
  red[tid] = mx; __syncthreads();
  for (int s = 128; s > 0; s >>= 1) { if (tid < s) red[tid] = fmaxf(red[tid], red[tid + s]); __syncthreads(); }
  mx = red[0]; __syncthreads();
  float sum = 0.0f;
  for (int j = tid; j < n; j += 256) sum += __expf(xr[j] - mx);
  red[tid] = sum; __syncthreads();
  for (int s = 128; s > 0; s >>= 1) { if (tid < s) red[tid] += red[tid + s]; __syncthreads(); }
  float lse = mx + logf(red[0]);
  for (int j = tid; j < n; j += 256) xr[j] -= lse;
}

// =======================================================================
extern "C" void kernel_launch(void* const* d_in, const int* in_sizes, int n_in,
                              void* d_out, int out_size, void* d_ws, size_t ws_size,
                              hipStream_t stream) {
  (void)in_sizes; (void)n_in; (void)out_size; (void)ws_size;
  const int*   tokens  = (const int*)  d_in[0];
  const float* EncOut  = (const float*)d_in[1];
  const float* emb     = (const float*)d_in[2];
  const float* fa1_w   = (const float*)d_in[3];
  const float* fa1_b   = (const float*)d_in[4];
  const float* fa2_w   = (const float*)d_in[5];
  const float* fa2_b   = (const float*)d_in[6];
  const float* fa3_w   = (const float*)d_in[7];
  const float* fa3_b   = (const float*)d_in[8];
  const float* in_w    = (const float*)d_in[9];
  const float* conv_w  = (const float*)d_in[10];
  const float* conv_b  = (const float*)d_in[11];
  const float* xp_w    = (const float*)d_in[12];
  const float* dt_w    = (const float*)d_in[13];
  const float* dt_b    = (const float*)d_in[14];
  const float* A_log   = (const float*)d_in[15];
  const float* Dp      = (const float*)d_in[16];
  const float* out_w   = (const float*)d_in[17];
  const float* norm_w  = (const float*)d_in[18];
  const float* normf_w = (const float*)d_in[19];
  float* out = (float*)d_out;

  // -------- workspace layout (256B aligned slabs) --------
  char* p = (char*)d_ws;
  auto alloc = [&](size_t bytes) -> char* {
    char* r = p; p += (bytes + 255) & ~(size_t)255; return r;
  };
  float*  enc    = (float*) alloc((size_t)Bn * Dn * 4);
  __bf16* fawb   = (__bf16*)alloc((size_t)1536 * 1024 * 2);
  float*  fab    = (float*) alloc((size_t)1536 * 4);
  __bf16* inwb   = (__bf16*)alloc((size_t)NLn * 2 * EDn * Dn * 2);
  __bf16* outwb  = (__bf16*)alloc((size_t)NLn * Dn * EDn * 2);
  __bf16* xpwb   = (__bf16*)alloc((size_t)NLn * 64 * EDn * 2);
  __bf16* dtwb   = (__bf16*)alloc((size_t)NLn * EDn * DTRn * 2);
  __bf16* embb   = (__bf16*)alloc((size_t)Vn * Dn * 2);
  float*  Aneg   = (float*) alloc((size_t)NLn * EDn * Nst * 4);
  __bf16* xe     = (__bf16*)alloc((size_t)Mrows * 1024 * 2);   // also reused as xi_bf16
  float*  hbuf   = (float*) alloc((size_t)Mrows * Dn * 4);
  __bf16* xn     = (__bf16*)alloc((size_t)Mrows * Dn * 2);
  float*  big0   = (float*) alloc((size_t)Mrows * 2048 * 4);   // gates / xz
  float*  xiconv = (float*) alloc((size_t)Mrows * EDn * 4);
  float*  dbc    = (float*) alloc((size_t)Mrows * 64 * 4);
  __bf16* dtb    = (__bf16*)alloc((size_t)Mrows * DTRn * 2);
  float*  delta  = (float*) alloc((size_t)Mrows * EDn * 4);
  __bf16* ygate  = (__bf16*)alloc((size_t)Mrows * EDn * 2);
  __bf16* xibf   = xe;   // xe dead once gates are computed

  auto nblk = [](size_t n) { return (unsigned)((n + 255) / 256); };

  // -------- setup: enc mean, bf16 weight conversion, A = -exp(A_log) --------
  enc_mean_kernel<<<nblk(Bn * Dn), 256, 0, stream>>>(EncOut, enc);
  cvt_bf16_kernel<<<nblk((size_t)Dn * 2 * Dn), 256, 0, stream>>>(fa1_w, fawb, (size_t)Dn * 2 * Dn);
  cvt_bf16_kernel<<<nblk((size_t)Dn * 2 * Dn), 256, 0, stream>>>(fa2_w, fawb + (size_t)Dn * 2 * Dn, (size_t)Dn * 2 * Dn);
  cvt_bf16_kernel<<<nblk((size_t)Dn * 2 * Dn), 256, 0, stream>>>(fa3_w, fawb + (size_t)2 * Dn * 2 * Dn, (size_t)Dn * 2 * Dn);
  copy_f32_kernel<<<nblk(Dn), 256, 0, stream>>>(fa1_b, fab, Dn);
  copy_f32_kernel<<<nblk(Dn), 256, 0, stream>>>(fa2_b, fab + Dn, Dn);
  copy_f32_kernel<<<nblk(Dn), 256, 0, stream>>>(fa3_b, fab + 2 * Dn, Dn);
  cvt_bf16_kernel<<<nblk((size_t)NLn * 2 * EDn * Dn), 256, 0, stream>>>(in_w, inwb, (size_t)NLn * 2 * EDn * Dn);
  cvt_bf16_kernel<<<nblk((size_t)NLn * Dn * EDn), 256, 0, stream>>>(out_w, outwb, (size_t)NLn * Dn * EDn);
  cvt_bf16_kernel<<<nblk((size_t)NLn * 64 * EDn), 256, 0, stream>>>(xp_w, xpwb, (size_t)NLn * 64 * EDn);
  cvt_bf16_kernel<<<nblk((size_t)NLn * EDn * DTRn), 256, 0, stream>>>(dt_w, dtwb, (size_t)NLn * EDn * DTRn);
  cvt_bf16_kernel<<<nblk((size_t)Vn * Dn), 256, 0, stream>>>(emb, embb, (size_t)Vn * Dn);
  neg_exp_kernel<<<nblk((size_t)NLn * EDn * Nst), 256, 0, stream>>>(A_log, Aneg, (size_t)NLn * EDn * Nst);

  // -------- embedding+enc concat, 3 fused sigmoid-gate GEMMs, combine --------
  build_xe_kernel<<<nblk((size_t)Mrows * Dn), 256, 0, stream>>>(tokens, emb, enc, xe);
  gemm_wmma_bf16<float, EPI_SIGB><<<dim3(1536 / 128, Mrows / 128), 256, 0, stream>>>(
      xe, fawb, big0, fab, nullptr, Mrows, 1536, 1024);
  gates_combine_kernel<<<nblk((size_t)Mrows * Dn), 256, 0, stream>>>(big0, enc, hbuf);

  // -------- 4 mamba layers --------
  for (int i = 0; i < NLn; ++i) {
    rmsnorm_kernel<<<Mrows, 256, 0, stream>>>(hbuf, norm_w + (size_t)i * Dn, xn);
    gemm_wmma_bf16<float, EPI_NONE><<<dim3(2048 / 128, Mrows / 128), 256, 0, stream>>>(
        xn, inwb + (size_t)i * 2 * EDn * Dn, big0, nullptr, nullptr, Mrows, 2048, Dn);
    conv_silu_kernel<<<nblk((size_t)Mrows * EDn), 256, 0, stream>>>(
        big0, conv_w + (size_t)i * EDn * 4, conv_b + (size_t)i * EDn, xiconv, xibf);
    gemm_wmma_bf16<float, EPI_NONE><<<dim3(1, Mrows / 128), 256, 0, stream>>>(
        xibf, xpwb + (size_t)i * 64 * EDn, dbc, nullptr, nullptr, Mrows, 64, EDn);
    slice_cvt_kernel<<<nblk((size_t)Mrows * DTRn), 256, 0, stream>>>(dbc, dtb, Mrows, 64, DTRn);
    gemm_wmma_bf16<float, EPI_SOFTPLUSB><<<dim3(EDn / 128, Mrows / 128), 256, 0, stream>>>(
        dtb, dtwb + (size_t)i * EDn * DTRn, delta, dt_b + (size_t)i * EDn, nullptr, Mrows, EDn, DTRn);
    scan_kernel<<<Bn, 1024, 0, stream>>>(
        dbc, delta, xiconv, big0, Aneg + (size_t)i * EDn * Nst, Dp + (size_t)i * EDn, ygate);
    gemm_wmma_bf16<float, EPI_RESID><<<dim3(Dn / 128, Mrows / 128), 256, 0, stream>>>(
        ygate, outwb + (size_t)i * Dn * EDn, hbuf, nullptr, hbuf, Mrows, Dn, EDn);
  }

  // -------- final norm, logits GEMM straight into d_out, log_softmax in place --------
  rmsnorm_kernel<<<Mrows, 256, 0, stream>>>(hbuf, normf_w, xn);
  gemm_wmma_bf16<float, EPI_NONE><<<dim3((Vn + 127) / 128, Mrows / 128), 256, 0, stream>>>(
      xn, embb, out, nullptr, nullptr, Mrows, Vn, Dn);
  logsoftmax_kernel<<<Mrows, 256, 0, stream>>>(out, Vn);
}
